// Model_68152541053064
// MI455X (gfx1250) — matrix operand order repaired and checked by reading, not compiled, not run on hardware
//
#include <hip/hip_runtime.h>
#include <hip/hip_bf16.h>
#include <math.h>

// ---------------- CDNA5 WMMA types/helpers (gfx1250, wave32) ----------------
typedef __attribute__((ext_vector_type(16))) __bf16 v16bf;
typedef __attribute__((ext_vector_type(8)))  __bf16 v8bf;
typedef __attribute__((ext_vector_type(8)))  float  v8f;

#define D_MODEL 512
#define SEQ     1024
#define S_TOT   1536
#define NB      4
#define NH      8
#define DKH     64

// B-fragment (32x16, 16-bit): lane (g=lane/16, n=lane%16) holds B[g*16+h][n],
// h=0..15 -> 16 contiguous elements when the contraction dim is contiguous.
__device__ __forceinline__ v16bf ldB(const __bf16* p) { return *(const v16bf*)p; }

// A-fragment (16x32, 16-bit): lane holds A[m=lane%16][k0+g*16+h]:
// 16 contiguous elements at p[0..15], matching the B-fragment k order.
__device__ __forceinline__ v16bf ldA(const __bf16* p) {
  return *(const v16bf*)p;
}

__device__ __forceinline__ v8f wmma_bf16(v16bf a, v16bf b, v8f c) {
  return __builtin_amdgcn_wmma_f32_16x16x32_bf16(false, a, false, b, (short)0, c,
                                                 false, false);
}

// ---------------- generic bf16 GEMM: C = A(MxK) * W(NxK)^T (+bias)(+gelu) ---
// block = 256 threads = 8 waves; wave -> 32 rows x 64 cols (2 A-frags x 4 B
// tiles = 8 independent WMMA chains, W fragment reused twice); block -> 256x64.
__global__ void __launch_bounds__(256)
gemm_bf16(const __bf16* __restrict__ A, const __bf16* __restrict__ W,
          const float* __restrict__ bias, float* __restrict__ outF,
          __bf16* __restrict__ outB, int M, int N, int K, int gelu) {
  int wave = threadIdx.x >> 5, lane = threadIdx.x & 31;
  int n16 = lane & 15, g = lane >> 4;
  int rowBase = blockIdx.x * 256 + wave * 32;
  int colBase = blockIdx.y * 64;

  v8f acc[2][4];
#pragma unroll
  for (int rt = 0; rt < 2; ++rt)
#pragma unroll
    for (int f = 0; f < 4; ++f)
#pragma unroll
      for (int i = 0; i < 8; ++i) acc[rt][f][i] = 0.f;

  const __bf16* arow0 = A + (size_t)(rowBase + n16) * K;
  const __bf16* arow1 = arow0 + (size_t)16 * K;
  for (int k0 = 0; k0 < K; k0 += 32) {
    if (k0 + 64 < K) {  // stream A ahead through WGP cache (global_prefetch_b8)
      __builtin_prefetch(arow0 + k0 + 64, 0, 1);
      __builtin_prefetch(arow1 + k0 + 64, 0, 1);
    }
    v16bf a0 = ldA(arow0 + k0 + g * 16);
    v16bf a1 = ldA(arow1 + k0 + g * 16);
#pragma unroll
    for (int f = 0; f < 4; ++f) {
      const __bf16* wrow =
          W + (size_t)(colBase + f * 16 + n16) * K + k0 + g * 16;
      v16bf bf = ldB(wrow);
      acc[0][f] = wmma_bf16(a0, bf, acc[0][f]);
      acc[1][f] = wmma_bf16(a1, bf, acc[1][f]);
    }
  }
#pragma unroll
  for (int rt = 0; rt < 2; ++rt)
#pragma unroll
    for (int f = 0; f < 4; ++f) {
      int col = colBase + f * 16 + n16;
      float bv = bias ? bias[col] : 0.f;
#pragma unroll
      for (int r = 0; r < 8; ++r) {
        int row = rowBase + rt * 16 + g * 8 + r;  // C layout: M = g*8+r, N = n16
        float v = acc[rt][f][r] + bv;
        if (gelu) v = 0.5f * v * (1.f + erff(v * 0.70710678118f));
        size_t idx = (size_t)row * N + col;
        if (outF) outF[idx] = v;
        if (outB) outB[idx] = (__bf16)v;
      }
    }
}

// ---------------- flash attention (bf16 WMMA, online softmax) ---------------
// grid (B*H, S/128), block 128 = 4 waves; each wave owns TWO 16-query tiles
// (32 queries) so K/V B-fragments are loaded once and used twice.
__global__ void __launch_bounds__(128)
attn_flash(const __bf16* __restrict__ q, const __bf16* __restrict__ k,
           const __bf16* __restrict__ vt,  // (B*H, 64, S) = V transposed
           const unsigned char* __restrict__ mask, __bf16* __restrict__ o) {
  __shared__ __align__(32) __bf16 plds[4][2][16 * 32];
  int wave = threadIdx.x >> 5, lane = threadIdx.x & 31;
  int n16 = lane & 15, g = lane >> 4;
  int bh = blockIdx.x, b = bh >> 3, hd = bh & 7;
  int qbase = (blockIdx.y * 4 + wave) * 32;
  const float scale = 0.125f;  // 1/sqrt(64)

  // Q A-fragments: 2 query tiles x 2 K-steps of DK=64 (reused for all chunks)
  v16bf aq[2][2];
#pragma unroll
  for (int qt = 0; qt < 2; ++qt) {
    const __bf16* qrow =
        q + ((size_t)b * S_TOT + qbase + qt * 16 + n16) * D_MODEL + hd * DKH;
    aq[qt][0] = ldA(qrow + g * 16);
    aq[qt][1] = ldA(qrow + 32 + g * 16);
  }

  float rmax[2][8], rsum[2][8];
  v8f O[2][4];
#pragma unroll
  for (int qt = 0; qt < 2; ++qt) {
#pragma unroll
    for (int r = 0; r < 8; ++r) { rmax[qt][r] = -3.0e38f; rsum[qt][r] = 0.f; }
#pragma unroll
    for (int f = 0; f < 4; ++f)
#pragma unroll
      for (int i = 0; i < 8; ++i) O[qt][f][i] = 0.f;
  }

  for (int kb = 0; kb < S_TOT; kb += 32) {
    v8f sc[2][2];
#pragma unroll
    for (int qt = 0; qt < 2; ++qt)
#pragma unroll
      for (int t = 0; t < 2; ++t)
#pragma unroll
        for (int i = 0; i < 8; ++i) sc[qt][t][i] = 0.f;

    // scores = Q * K^T : B-frag b[h] = K[key n][feat s*32+g*16+h] (contiguous)
#pragma unroll
    for (int s = 0; s < 2; ++s) {
      const __bf16* kr0 =
          k + ((size_t)b * S_TOT + kb + n16) * D_MODEL + hd * DKH + s * 32 +
          g * 16;
      v16bf b0 = ldB(kr0);
      v16bf b1 = ldB(kr0 + (size_t)16 * D_MODEL);
#pragma unroll
      for (int qt = 0; qt < 2; ++qt) {
        sc[qt][0] = wmma_bf16(aq[qt][s], b0, sc[qt][0]);
        sc[qt][1] = wmma_bf16(aq[qt][s], b1, sc[qt][1]);
      }
    }
    // mask + scale + online softmax; row q = qbase+qt*16+g*8+r spans the
    // 16-lane column group, so reductions are xor-1/2/4/8 butterflies.
#pragma unroll
    for (int qt = 0; qt < 2; ++qt) {
#pragma unroll
      for (int r = 0; r < 8; ++r) {
        const unsigned char* m =
            mask + (size_t)(qbase + qt * 16 + g * 8 + r) * S_TOT + kb;
        float v0 = sc[qt][0][r] * scale; if (m[n16])      v0 = -1.0e9f;
        float v1 = sc[qt][1][r] * scale; if (m[16 + n16]) v1 = -1.0e9f;
        float mx = fmaxf(v0, v1);
#pragma unroll
        for (int off = 1; off < 16; off <<= 1)
          mx = fmaxf(mx, __shfl_xor(mx, off, 32));
        float nm = fmaxf(rmax[qt][r], mx);
        float alpha = __expf(rmax[qt][r] - nm);
        rmax[qt][r] = nm;
        float p0 = __expf(v0 - nm), p1 = __expf(v1 - nm);
        float ps = p0 + p1;
#pragma unroll
        for (int off = 1; off < 16; off <<= 1) ps += __shfl_xor(ps, off, 32);
        rsum[qt][r] = rsum[qt][r] * alpha + ps;
#pragma unroll
        for (int f = 0; f < 4; ++f) O[qt][f][r] *= alpha;
        // stash probs in C-layout; reread below in A-layout (LDS transpose)
        plds[wave][qt][(g * 8 + r) * 32 + n16]      = (__bf16)p0;
        plds[wave][qt][(g * 8 + r) * 32 + 16 + n16] = (__bf16)p1;
      }
    }
    asm volatile("s_wait_dscnt 0" ::: "memory");  // DS ops in-order per wave
    v16bf pa0 = ldA(&plds[wave][0][n16 * 32 + g * 16]);
    v16bf pa1 = ldA(&plds[wave][1][n16 * 32 + g * 16]);
    // O += P * V : B-frag from V^T rows (contiguous along keys), used twice
#pragma unroll
    for (int f = 0; f < 4; ++f) {
      const __bf16* vr =
          vt + ((size_t)bh * DKH + f * 16 + n16) * S_TOT + kb + g * 16;
      v16bf bv = ldB(vr);
      O[0][f] = wmma_bf16(pa0, bv, O[0][f]);
      O[1][f] = wmma_bf16(pa1, bv, O[1][f]);
    }
  }
#pragma unroll
  for (int qt = 0; qt < 2; ++qt)
#pragma unroll
    for (int f = 0; f < 4; ++f)
#pragma unroll
      for (int r = 0; r < 8; ++r) {
        int row = qbase + qt * 16 + g * 8 + r;
        float v = O[qt][f][r] / rsum[qt][r];
        o[((size_t)b * S_TOT + row) * D_MODEL + hd * DKH + f * 16 + n16] =
            (__bf16)v;
      }
}

// ---------------- small helper kernels --------------------------------------
__global__ void f32_to_bf16(const float* __restrict__ in,
                            __bf16* __restrict__ out, int n) {
  int i = blockIdx.x * blockDim.x + threadIdx.x;
  int stride = gridDim.x * blockDim.x;
  for (; i < n; i += stride) out[i] = (__bf16)in[i];
}

__global__ void transpose_v(const __bf16* __restrict__ v,
                            __bf16* __restrict__ vt, int n) {
  int i = blockIdx.x * blockDim.x + threadIdx.x;
  int stride = gridDim.x * blockDim.x;
  for (; i < n; i += stride) {
    int s = i % S_TOT;
    int rest = i / S_TOT;
    int d = rest & 63, bh = rest >> 6;
    int b = bh >> 3, hd = bh & 7;
    vt[i] = v[((size_t)b * S_TOT + s) * D_MODEL + hd * DKH + d];
  }
}

// token conv (circular pad, k=3) + temporal embeds + sinusoidal pos
__global__ void __launch_bounds__(256)
embed_kernel(const float* __restrict__ x_enc, const int* __restrict__ x_mark,
             const float* __restrict__ tok_W, const float* __restrict__ mE,
             const float* __restrict__ dE, const float* __restrict__ wE,
             const float* __restrict__ hE, float* __restrict__ seqf) {
  __shared__ float xv[21];
  __shared__ int mk[4];
  int bl = blockIdx.x, b = bl >> 10, l = bl & 1023;
  int t = threadIdx.x;
  if (t < 21) {
    int c = t / 3, kk = t % 3;
    int i = l + kk - 1;
    if (i < 0) i = SEQ - 1; else if (i >= SEQ) i = 0;
    xv[t] = x_enc[((size_t)b * SEQ + i) * 7 + c];
  }
  if (t >= 32 && t < 36) mk[t - 32] = x_mark[((size_t)b * SEQ + l) * 4 + (t - 32)];
  __syncthreads();
  const float kc = -9.210340371976184f / 512.f;  // -ln(10000)/D
#pragma unroll
  for (int e = 0; e < 2; ++e) {
    int d = 2 * t + e;
    float tok = 0.f;
#pragma unroll
    for (int c = 0; c < 7; ++c)
#pragma unroll
      for (int kk = 0; kk < 3; ++kk)
        tok += xv[c * 3 + kk] * tok_W[(size_t)d * 21 + c * 3 + kk];
    float tmp = mE[mk[0] * 512 + d] + dE[mk[1] * 512 + d] +
                wE[mk[2] * 512 + d] + hE[mk[3] * 512 + d];
    float div = __expf((float)(d & ~1) * kc);
    float pos = (e == 0) ? __sinf((float)l * div) : __cosf((float)l * div);
    seqf[((size_t)b * S_TOT + l) * 512 + d] = tok + tmp + pos;
  }
}

__global__ void __launch_bounds__(64)
down_kernel(const float* __restrict__ seqf, const float* __restrict__ down_W,
            const float* __restrict__ down_b, float* __restrict__ t1) {
  __shared__ float row[512];
  int bl = blockIdx.x, b = bl >> 10, l = bl & 1023;
  int t = threadIdx.x;
  const float* sr = seqf + ((size_t)b * S_TOT + l) * 512;
  for (int i = t; i < 512; i += 64) row[i] = sr[i];
  __syncthreads();
  float acc = down_b[t];
  const float* w = down_W + (size_t)t * 512;
#pragma unroll 4
  for (int kk = 0; kk < 512; ++kk) acc += row[kk] * w[kk];
  t1[((size_t)b * SEQ + l) * 64 + t] = acc;
}

__global__ void __launch_bounds__(256)
convup_kernel(const float* __restrict__ t1, const float* __restrict__ conv_W,
              const float* __restrict__ conv_b, const float* __restrict__ bn_g,
              const float* __restrict__ bn_b, const float* __restrict__ bn_m,
              const float* __restrict__ bn_v, const float* __restrict__ up_W,
              const float* __restrict__ up_b, float* __restrict__ seqf) {
  __shared__ float tl[128];
  __shared__ float hsh[64];
  int bj = blockIdx.x, b = bj >> 9, j = bj & 511;
  int t = threadIdx.x;
  if (t < 128) tl[t] = t1[((size_t)b * SEQ + 2 * j + (t >> 6)) * 64 + (t & 63)];
  __syncthreads();
  if (t < 64) {
    float acc = conv_b[t];
#pragma unroll 4
    for (int i = 0; i < 64; ++i) {
      acc += tl[i]      * conv_W[((size_t)t * 64 + i) * 2 + 0];
      acc += tl[64 + i] * conv_W[((size_t)t * 64 + i) * 2 + 1];
    }
    acc = (acc - bn_m[t]) * rsqrtf(bn_v[t] + 1e-5f) * bn_g[t] + bn_b[t];
    hsh[t] = acc > 0.f ? acc : (__expf(acc) - 1.f);  // elu
  }
  __syncthreads();
#pragma unroll
  for (int e = 0; e < 2; ++e) {
    int d = t + e * 256;
    float acc = up_b[d];
#pragma unroll 4
    for (int i = 0; i < 64; ++i) acc += hsh[i] * up_W[(size_t)d * 64 + i];
    seqf[((size_t)b * S_TOT + SEQ + j) * 512 + d] = acc;
  }
}

// layernorm over D=512, optional residual add; writes f32 + bf16 mirrors
__global__ void __launch_bounds__(256)
ln_kernel(const float* __restrict__ x, const float* __restrict__ res,
          const float* __restrict__ gamma, const float* __restrict__ beta,
          float eps, float* __restrict__ outF, __bf16* __restrict__ outB) {
  __shared__ float s1[256], s2[256];
  int row = blockIdx.x, t = threadIdx.x;
  const float* xr = x + (size_t)row * 512;
  const float* rr = res ? res + (size_t)row * 512 : nullptr;
  float a = xr[t] + (rr ? rr[t] : 0.f);
  float c = xr[t + 256] + (rr ? rr[t + 256] : 0.f);
  s1[t] = a + c;
  s2[t] = a * a + c * c;
  __syncthreads();
  for (int off = 128; off > 0; off >>= 1) {
    if (t < off) { s1[t] += s1[t + off]; s2[t] += s2[t + off]; }
    __syncthreads();
  }
  float mean = s1[0] * (1.f / 512.f);
  float var = s2[0] * (1.f / 512.f) - mean * mean;
  float inv = rsqrtf(var + eps);
  float y0 = (a - mean) * inv * gamma[t] + beta[t];
  float y1 = (c - mean) * inv * gamma[t + 256] + beta[t + 256];
  size_t i0 = (size_t)row * 512 + t;
  outF[i0] = y0;          outF[i0 + 256] = y1;
  outB[i0] = (__bf16)y0;  outB[i0 + 256] = (__bf16)y1;
}

// gather last pyramid position + final projection (tiny: 4x672x1024)
__global__ void __launch_bounds__(672)
pred_kernel(const float* __restrict__ seqf, const int* __restrict__ gidx,
            const float* __restrict__ pred_W, float* __restrict__ out) {
  __shared__ float last[1024];
  int b = blockIdx.x, t = threadIdx.x;
  int r0 = gidx[2046], r1 = gidx[2047];  // node (L-1) at both pyramid levels
  for (int i = t; i < 1024; i += 672)
    last[i] = (i < 512) ? seqf[((size_t)b * S_TOT + r0) * 512 + i]
                        : seqf[((size_t)b * S_TOT + r1) * 512 + (i - 512)];
  __syncthreads();
  float acc = 0.f;
  const float* w = pred_W + (size_t)t * 1024;
#pragma unroll 4
  for (int kk = 0; kk < 1024; ++kk) acc += last[kk] * w[kk];
  out[(size_t)b * 672 + t] = acc;
}

// ---------------- launch ----------------------------------------------------
extern "C" void kernel_launch(void* const* d_in, const int* in_sizes, int n_in,
                              void* d_out, int out_size, void* d_ws,
                              size_t ws_size, hipStream_t stream) {
  const float* x_enc   = (const float*)d_in[0];
  const int*   x_mark  = (const int*)d_in[1];
  const unsigned char* mask = (const unsigned char*)d_in[2];
  const int*   gidx    = (const int*)d_in[3];
  const float* tok_W   = (const float*)d_in[4];
  const float* month_E = (const float*)d_in[5];
  const float* day_E   = (const float*)d_in[6];
  const float* wday_E  = (const float*)d_in[7];
  const float* hour_E  = (const float*)d_in[8];
  const float* down_W  = (const float*)d_in[9];
  const float* down_b  = (const float*)d_in[10];
  const float* conv_W  = (const float*)d_in[11];
  const float* conv_b  = (const float*)d_in[12];
  const float* bn_g    = (const float*)d_in[13];
  const float* bn_b    = (const float*)d_in[14];
  const float* bn_m    = (const float*)d_in[15];
  const float* bn_v    = (const float*)d_in[16];
  const float* up_W    = (const float*)d_in[17];
  const float* up_b    = (const float*)d_in[18];
  const float* bot_g   = (const float*)d_in[19];
  const float* bot_b   = (const float*)d_in[20];
  const float* Wq      = (const float*)d_in[21];
  const float* Wk      = (const float*)d_in[22];
  const float* Wv      = (const float*)d_in[23];
  const float* fc_W    = (const float*)d_in[24];
  const float* fc_b    = (const float*)d_in[25];
  const float* a_g     = (const float*)d_in[26];
  const float* a_b     = (const float*)d_in[27];
  const float* w1_W    = (const float*)d_in[28];
  const float* w1_b    = (const float*)d_in[29];
  const float* w2_W    = (const float*)d_in[30];
  const float* w2_b    = (const float*)d_in[31];
  const float* f_g     = (const float*)d_in[32];
  const float* f_b     = (const float*)d_in[33];
  const float* pred_W  = (const float*)d_in[34];

  char* p = (char*)d_ws;
  auto alloc = [&](size_t bytes) -> char* {
    char* r = p;
    p += (bytes + 255) & ~(size_t)255;
    return r;
  };
  const size_t Mrows = (size_t)NB * S_TOT;  // 6144
  const size_t actF = Mrows * 512 * sizeof(float);
  const size_t actB = Mrows * 512 * sizeof(__bf16);
  const size_t wB   = (size_t)2 * 512 * 512 * sizeof(__bf16);

  float*  seqf = (float*)alloc(actF);
  __bf16* seqb = (__bf16*)alloc(actB);
  __bf16* qb   = (__bf16*)alloc(actB);
  __bf16* kb   = (__bf16*)alloc(actB);
  __bf16* vb   = (__bf16*)alloc(actB);
  __bf16* vtb  = (__bf16*)alloc(actB);
  __bf16* ob   = (__bf16*)alloc(actB);
  __bf16* hb   = (__bf16*)alloc(actB);
  float*  tmpf = (float*)alloc(actF);
  float*  t1   = (float*)alloc((size_t)NB * SEQ * 64 * sizeof(float));
  __bf16* cWq  = (__bf16*)alloc(wB);
  __bf16* cWk  = (__bf16*)alloc(wB);
  __bf16* cWv  = (__bf16*)alloc(wB);
  __bf16* cFc  = (__bf16*)alloc(wB);
  __bf16* cW1  = (__bf16*)alloc(wB);
  __bf16* cW2  = (__bf16*)alloc(wB);

  const int wn = 2 * 512 * 512;
  f32_to_bf16<<<1024, 256, 0, stream>>>(Wq,   cWq, wn);
  f32_to_bf16<<<1024, 256, 0, stream>>>(Wk,   cWk, wn);
  f32_to_bf16<<<1024, 256, 0, stream>>>(Wv,   cWv, wn);
  f32_to_bf16<<<1024, 256, 0, stream>>>(fc_W, cFc, wn);
  f32_to_bf16<<<1024, 256, 0, stream>>>(w1_W, cW1, wn);
  f32_to_bf16<<<1024, 256, 0, stream>>>(w2_W, cW2, wn);

  embed_kernel<<<NB * SEQ, 256, 0, stream>>>(x_enc, x_mark, tok_W, month_E,
                                             day_E, wday_E, hour_E, seqf);
  down_kernel<<<NB * SEQ, 64, 0, stream>>>(seqf, down_W, down_b, t1);
  convup_kernel<<<NB * 512, 256, 0, stream>>>(t1, conv_W, conv_b, bn_g, bn_b,
                                              bn_m, bn_v, up_W, up_b, seqf);
  ln_kernel<<<(int)Mrows, 256, 0, stream>>>(seqf, nullptr, bot_g, bot_b,
                                            1e-5f, seqf, seqb);

  dim3 gg(24, 8);  // 6144/256 x 512/64
  for (int i = 0; i < 2; ++i) {
    size_t wo = (size_t)i * 512 * 512;
    gemm_bf16<<<gg, 256, 0, stream>>>(seqb, cWq + wo, nullptr, nullptr, qb,
                                      6144, 512, 512, 0);
    gemm_bf16<<<gg, 256, 0, stream>>>(seqb, cWk + wo, nullptr, nullptr, kb,
                                      6144, 512, 512, 0);
    gemm_bf16<<<gg, 256, 0, stream>>>(seqb, cWv + wo, nullptr, nullptr, vb,
                                      6144, 512, 512, 0);
    transpose_v<<<1024, 256, 0, stream>>>(vb, vtb, NB * NH * DKH * S_TOT);
    attn_flash<<<dim3(NB * NH, S_TOT / 128), 128, 0, stream>>>(qb, kb, vtb,
                                                               mask, ob);
    gemm_bf16<<<gg, 256, 0, stream>>>(ob, cFc + wo, fc_b + i * 512, tmpf,
                                      nullptr, 6144, 512, 512, 0);
    ln_kernel<<<(int)Mrows, 256, 0, stream>>>(tmpf, seqf, a_g + i * 512,
                                              a_b + i * 512, 1e-6f, seqf, seqb);
    gemm_bf16<<<gg, 256, 0, stream>>>(seqb, cW1 + wo, w1_b + i * 512, nullptr,
                                      hb, 6144, 512, 512, 1);
    gemm_bf16<<<gg, 256, 0, stream>>>(hb, cW2 + wo, w2_b + i * 512, tmpf,
                                      nullptr, 6144, 512, 512, 0);
    ln_kernel<<<(int)Mrows, 256, 0, stream>>>(tmpf, seqf, f_g + i * 512,
                                              f_b + i * 512, 1e-6f, seqf, seqb);
  }
  pred_kernel<<<NB, 672, 0, stream>>>(seqf, gidx, pred_W, (float*)d_out);
}
